// RNNEncoderDecoder_64965675319867
// MI455X (gfx1250) — matrix operand order repaired and checked by reading, not compiled, not run on hardware
//
#include <hip/hip_runtime.h>

typedef __attribute__((ext_vector_type(16))) __bf16 v16bf;
typedef __attribute__((ext_vector_type(8)))  float  v8f;
typedef __attribute__((ext_vector_type(4)))  unsigned int v4u32;
typedef __attribute__((ext_vector_type(8)))  int v8i32;
typedef __attribute__((ext_vector_type(4)))  int v4i32;

#define B_  256
#define T_  512
#define IN_ 128
#define H_  64

// ---- workspace byte offsets (all 256B aligned) ----
#define XW_BYTES (16ul * 512ul * 4ul * 256ul * 4ul)   // 32 MB of xw_enc fragments
#define OFF_XW   0ul
#define OFF_WIHE (XW_BYTES)                            // 64x128 bf16
#define OFF_WHHE (OFF_WIHE + 64ul*128ul*2ul)           // 64x64 bf16
#define OFF_WIHD (OFF_WHHE + 64ul*64ul*2ul)            // 3 contiguous 64x64 bf16:
#define OFF_WHHD (OFF_WIHD + 64ul*64ul*2ul)            //   W_ih_dec | W_hh_dec | W_out
#define OFF_WOUT (OFF_WHHD + 64ul*64ul*2ul)
#define OFF_BE   (OFF_WOUT + 64ul*64ul*2ul)            // 64 f32 (b_ih+b_hh enc)
#define OFF_BD   (OFF_BE + 256ul)                      // 64 f32 (dec)

__device__ __forceinline__ v8f wmma_bf16(v16bf a, v16bf b, v8f c) {
  return __builtin_amdgcn_wmma_f32_16x16x32_bf16(false, a, false, b, (short)0, c,
                                                 false, false);
}

// tanh(x) = 1 - 2/(exp(2x)+1); exact limits at +-inf via rcp(inf)=0.
__device__ __forceinline__ float fast_tanh(float x) {
  float e = __expf(2.0f * x);
  return 1.0f - 2.0f * __builtin_amdgcn_rcpf(e + 1.0f);
}

// Load 8 consecutive f32 and place as bf16 into elements [base..base+7].
__device__ __forceinline__ void pack8(const float* p, v16bf& d, int base) {
  float4 a = *(const float4*)p;
  float4 b = *(const float4*)(p + 4);
  d[base + 0] = (__bf16)a.x; d[base + 1] = (__bf16)a.y;
  d[base + 2] = (__bf16)a.z; d[base + 3] = (__bf16)a.w;
  d[base + 4] = (__bf16)b.x; d[base + 5] = (__bf16)b.y;
  d[base + 6] = (__bf16)b.z; d[base + 7] = (__bf16)b.w;
}

__device__ __forceinline__ v8f load8f(const float* p) {
  float4 a = *(const float4*)p;
  float4 b = *(const float4*)(p + 4);
  v8f r;
  r[0] = a.x; r[1] = a.y; r[2] = a.z; r[3] = a.w;
  r[4] = b.x; r[5] = b.y; r[6] = b.z; r[7] = b.w;
  return r;
}

__device__ __forceinline__ void store8f(float* p, v8f a) {
  *(float4*)p       = make_float4(a[0], a[1], a[2], a[3]);
  *(float4*)(p + 4) = make_float4(a[4], a[5], a[6], a[7]);
}

// TDM: 1-D DMA of `nelem` 4-byte elements from global to LDS (6-arg builtin:
// g0, g1, g2, g3, g_extra, cpol). Descriptor bitfields per CDNA5 ISA ch.8.
__device__ __forceinline__ void tdm_load_1d(unsigned int lds_byte_addr,
                                            const void* gptr,
                                            unsigned int nelem) {
  unsigned long long ga = (unsigned long long)(const char*)gptr;
  v4u32 g0;
  g0[0] = 1u;                                   // count=1, user descriptor
  g0[1] = lds_byte_addr;                        // lds_addr (bytes)
  g0[2] = (unsigned int)ga;                     // global_addr[31:0]
  g0[3] = (unsigned int)(ga >> 32) | (2u << 30);// global_addr[56:32] | type=2
  v8i32 g1;
  g1[0] = (int)(2u << 16);                      // data_size = 2 (4 bytes)
  g1[1] = (int)((nelem & 0xFFFFu) << 16);       // tensor_dim0[15:0]
  g1[2] = (int)(((nelem >> 16) & 0xFFFFu)       // tensor_dim0[31:16]
                | (1u << 16));                  // tensor_dim1 = 1
  g1[3] = (int)((nelem & 0xFFFFu) << 16);       // tile_dim0 = nelem (1-D tile)
  g1[4] = 0;                                    // tile_dim1/2 unused
  g1[5] = (int)nelem;                           // tensor_dim0_stride[31:0]
  g1[6] = 0;
  g1[7] = 0;
  v4i32 z4 = {0, 0, 0, 0};
  v8i32 z8 = {0, 0, 0, 0, 0, 0, 0, 0};
  __builtin_amdgcn_tensor_load_to_lds(g0, g1, z4, z4, z8, 0);
}

// ---- Kernel 0: convert weights to bf16, fold bias pairs ----
__global__ void k_prep(const float* wihE, const float* whhE,
                       const float* bihE, const float* bhhE,
                       const float* wihD, const float* whhD,
                       const float* bihD, const float* bhhD,
                       const float* wout, unsigned char* ws) {
  int i = blockIdx.x * blockDim.x + threadIdx.x;
  __bf16* oIHE = (__bf16*)(ws + OFF_WIHE);
  __bf16* oHHE = (__bf16*)(ws + OFF_WHHE);
  __bf16* oIHD = (__bf16*)(ws + OFF_WIHD);
  __bf16* oHHD = (__bf16*)(ws + OFF_WHHD);
  __bf16* oOUT = (__bf16*)(ws + OFF_WOUT);
  float* bE = (float*)(ws + OFF_BE);
  float* bD = (float*)(ws + OFF_BD);
  if (i < 64 * 128) oIHE[i] = (__bf16)wihE[i];
  if (i < 64 * 64) {
    oHHE[i] = (__bf16)whhE[i];
    oIHD[i] = (__bf16)wihD[i];
    oHHD[i] = (__bf16)whhD[i];
    oOUT[i] = (__bf16)wout[i];
  }
  if (i < 64) {
    bE[i] = bihE[i] + bhhE[i];
    bD[i] = bihD[i] + bhhD[i];
  }
}

// ---- Kernel 1: xw_enc = seq @ W_ih_enc^T + bias, written in C-fragment layout.
__global__ void __launch_bounds__(128) k_enc_inproj(
    const float* __restrict__ seq, const unsigned char* __restrict__ wsb,
    float* __restrict__ xw) {
  const __bf16* wih  = (const __bf16*)(wsb + OFF_WIHE);
  const float* biasE = (const float*)(wsb + OFF_BE);
  int wid = blockIdx.x * 4 + (threadIdx.x >> 5);
  int l = threadIdx.x & 31, lr = l & 15, lh = l >> 4;
  int mt = wid >> 9, t = wid & 511;

  v8f acc[4];
#pragma unroll
  for (int j = 0; j < 4; ++j) {
    float bv = biasE[16 * j + lr];
#pragma unroll
    for (int v = 0; v < 8; ++v) acc[j][v] = bv;
  }

  const float* arow = seq + ((size_t)(mt * 16 + lr) * T_ + t) * IN_;
#pragma unroll
  for (int s = 0; s < 4; ++s) {
    int k0 = 32 * s;
    v16bf A;
    pack8(arow + k0 + lh * 16, A, 0);
    pack8(arow + k0 + lh * 16 + 8, A, 8);
#pragma unroll
    for (int j = 0; j < 4; ++j) {
      v16bf Bm = *(const v16bf*)(wih + (16 * j + lr) * IN_ + k0 + lh * 16);
      acc[j] = wmma_bf16(A, Bm, acc[j]);
    }
  }

  float* fb = xw + ((size_t)(mt * T_ + t) * 4) * 256;
#pragma unroll
  for (int j = 0; j < 4; ++j) store8f(fb + j * 256 + (size_t)l * 8, acc[j]);
}

// ---- Kernel 2: fused enc-RNN -> dec input proj -> dec-RNN -> output proj.
// One wave per 16-batch tile. W_hh_enc pinned in VGPRs; W_ih_dec/W_hh_dec/W_out
// staged in LDS; xw fragments DMA'd by the Tensor Data Mover, double-buffered.
__global__ void __launch_bounds__(32) k_rnn_fused(
    const float* __restrict__ xw, const unsigned char* __restrict__ wsb,
    const float* __restrict__ bout, float* __restrict__ out) {
  __shared__ __align__(32) float  ldsE[16 * 64];
  __shared__ __align__(32) float  ldsD[16 * 64];
  __shared__ __align__(32) __bf16 ldsW[3 * 4096];  // WD | WH | WO (row-major)
  __shared__ __align__(32) float  ldsX[2][1024];   // TDM double buffer

  const __bf16* whhE  = (const __bf16*)(wsb + OFF_WHHE);
  const __bf16* wPack = (const __bf16*)(wsb + OFF_WIHD);  // 3 contiguous mats
  const float* biasD  = (const float*)(wsb + OFF_BD);

  int l = threadIdx.x, lr = l & 15, lh = l >> 4;
  int mt = blockIdx.x;

  // One-time stage of the 3 decoder-side matrices into LDS (24 KB bf16).
  {
    const uint4* src = (const uint4*)wPack;
    uint4* dst = (uint4*)ldsW;
    for (int i = l; i < (3 * 4096 * 2) / 16; i += 32) dst[i] = src[i];
    // Initialize the TDM destination buffers so no load is "from undef".
    float4* z = (float4*)&ldsX[0][0];
    for (int i = l; i < 2048 / 4; i += 32) z[i] = make_float4(0, 0, 0, 0);
  }
  __syncthreads();

  // Encoder recurrence weights: keep resident in VGPRs (critical path).
  v16bf WE[2][4];
#pragma unroll
  for (int s = 0; s < 2; ++s)
#pragma unroll
    for (int j = 0; j < 4; ++j)
      WE[s][j] = *(const v16bf*)(whhE + (16 * j + lr) * H_ + 32 * s + lh * 16);

  float bD[4], bO[4];
#pragma unroll
  for (int j = 0; j < 4; ++j) {
    bD[j] = biasD[16 * j + lr];
    bO[j] = bout[16 * j + lr];
  }

  v16bf hE[2], hD[2];  // hidden states in A-fragment layout (h0 = 0)
#pragma unroll
  for (int s = 0; s < 2; ++s)
#pragma unroll
    for (int e = 0; e < 16; ++e) {
      hE[s][e] = (__bf16)0.0f;
      hD[s][e] = (__bf16)0.0f;
    }

  const float* fb = xw + (size_t)mt * T_ * 1024;
  unsigned int xa[2];
  xa[0] = (unsigned int)(unsigned long long)(void*)&ldsX[0][0];
  xa[1] = (unsigned int)(unsigned long long)(void*)&ldsX[1][0];

  // Prime the pipeline: DMA timestep 0 into buffer 0.
  tdm_load_1d(xa[0], fb, 1024);
  __builtin_amdgcn_s_wait_tensorcnt(0);
  asm volatile("" ::: "memory");

  for (int t = 0; t < T_; ++t) {
    const float* cur = &ldsX[t & 1][0];
    if (t + 1 < T_)  // overlap next tile's DMA with this step's compute
      tdm_load_1d(xa[(t + 1) & 1], fb + (size_t)(t + 1) * 1024, 1024);

    // h_enc = tanh(xw_t + h_enc @ W_hh_enc^T)
    v8f aE[4];
#pragma unroll
    for (int j = 0; j < 4; ++j) aE[j] = load8f(cur + j * 256 + l * 8);
#pragma unroll
    for (int s = 0; s < 2; ++s)
#pragma unroll
      for (int j = 0; j < 4; ++j) aE[j] = wmma_bf16(hE[s], WE[s][j], aE[j]);
#pragma unroll
    for (int j = 0; j < 4; ++j)
#pragma unroll
      for (int v = 0; v < 8; ++v) aE[j][v] = fast_tanh(aE[j][v]);

    // C-layout -> LDS (row-major 16x64) -> A-layout bf16 fragments
#pragma unroll
    for (int j = 0; j < 4; ++j)
#pragma unroll
      for (int v = 0; v < 8; ++v)
        ldsE[(v + 8 * lh) * 64 + lr + 16 * j] = aE[j][v];
    __syncthreads();
#pragma unroll
    for (int s = 0; s < 2; ++s) {
      pack8(&ldsE[lr * 64 + 32 * s + lh * 16], hE[s], 0);
      pack8(&ldsE[lr * 64 + 32 * s + lh * 16 + 8], hE[s], 8);
    }

    // h_dec = tanh(h_enc @ W_ih_dec^T + bias_dec + h_dec @ W_hh_dec^T)
    v8f aD[4];
#pragma unroll
    for (int j = 0; j < 4; ++j)
#pragma unroll
      for (int v = 0; v < 8; ++v) aD[j][v] = bD[j];
#pragma unroll
    for (int s = 0; s < 2; ++s)
#pragma unroll
      for (int j = 0; j < 4; ++j) {
        v16bf WDf = *(const v16bf*)(ldsW + (16 * j + lr) * H_ + 32 * s + lh * 16);
        aD[j] = wmma_bf16(hE[s], WDf, aD[j]);
      }
#pragma unroll
    for (int s = 0; s < 2; ++s)
#pragma unroll
      for (int j = 0; j < 4; ++j) {
        v16bf WHf = *(const v16bf*)(ldsW + 4096 + (16 * j + lr) * H_ + 32 * s + lh * 16);
        aD[j] = wmma_bf16(hD[s], WHf, aD[j]);
      }
#pragma unroll
    for (int j = 0; j < 4; ++j)
#pragma unroll
      for (int v = 0; v < 8; ++v) aD[j][v] = fast_tanh(aD[j][v]);

#pragma unroll
    for (int j = 0; j < 4; ++j)
#pragma unroll
      for (int v = 0; v < 8; ++v)
        ldsD[(v + 8 * lh) * 64 + lr + 16 * j] = aD[j][v];
    __syncthreads();
#pragma unroll
    for (int s = 0; s < 2; ++s) {
      pack8(&ldsD[lr * 64 + 32 * s + lh * 16], hD[s], 0);
      pack8(&ldsD[lr * 64 + 32 * s + lh * 16 + 8], hD[s], 8);
    }

    // out_t = h_dec @ W_out^T + b_out
    v8f aO[4];
#pragma unroll
    for (int j = 0; j < 4; ++j)
#pragma unroll
      for (int v = 0; v < 8; ++v) aO[j][v] = bO[j];
#pragma unroll
    for (int s = 0; s < 2; ++s)
#pragma unroll
      for (int j = 0; j < 4; ++j) {
        v16bf WOf = *(const v16bf*)(ldsW + 8192 + (16 * j + lr) * H_ + 32 * s + lh * 16);
        aO[j] = wmma_bf16(hD[s], WOf, aO[j]);
      }

    size_t ob = ((size_t)(mt * 16) * T_ + t) * 64;
#pragma unroll
    for (int j = 0; j < 4; ++j)
#pragma unroll
      for (int v = 0; v < 8; ++v)
        out[ob + (size_t)(v + 8 * lh) * (T_ * 64) + lr + 16 * j] = aO[j][v];

    // Next iteration's xw tile must be resident before we read it.
    __builtin_amdgcn_s_wait_tensorcnt(0);
    asm volatile("" ::: "memory");
  }
}

extern "C" void kernel_launch(void* const* d_in, const int* in_sizes, int n_in,
                              void* d_out, int out_size, void* d_ws,
                              size_t ws_size, hipStream_t stream) {
  (void)in_sizes; (void)n_in; (void)out_size; (void)ws_size;
  const float* seq  = (const float*)d_in[0];
  const float* wihE = (const float*)d_in[1];
  const float* whhE = (const float*)d_in[2];
  const float* bihE = (const float*)d_in[3];
  const float* bhhE = (const float*)d_in[4];
  const float* wihD = (const float*)d_in[5];
  const float* whhD = (const float*)d_in[6];
  const float* bihD = (const float*)d_in[7];
  const float* bhhD = (const float*)d_in[8];
  const float* wout = (const float*)d_in[9];
  const float* bout = (const float*)d_in[10];
  unsigned char* ws = (unsigned char*)d_ws;
  float* xw  = (float*)(ws + OFF_XW);
  float* out = (float*)d_out;

  hipLaunchKernelGGL(k_prep, dim3(32), dim3(256), 0, stream, wihE, whhE, bihE,
                     bhhE, wihD, whhD, bihD, bhhD, wout, ws);
  hipLaunchKernelGGL(k_enc_inproj, dim3(2048), dim3(128), 0, stream, seq, ws,
                     xw);
  hipLaunchKernelGGL(k_rnn_fused, dim3(16), dim3(32), 0, stream, xw, ws, bout,
                     out);
}
